// ModelNet10ShapePrior_25074019074117
// MI455X (gfx1250) — matrix operand order repaired and checked by reading, not compiled, not run on hardware
//
#include <hip/hip_runtime.h>
#include <hip/hip_bf16.h>

typedef __bf16 bf16_t;
typedef __attribute__((ext_vector_type(16))) __bf16 v16bf;
typedef __attribute__((ext_vector_type(8)))  __bf16 v8bf;
typedef __attribute__((ext_vector_type(4)))  __bf16 v4bf;
typedef __attribute__((ext_vector_type(8)))  float  v8f;
typedef __attribute__((ext_vector_type(4)))  float  v4f;

#define Bc    32
#define Nc    8192
#define Hc    512
#define Hh    256
#define Ec    10
#define TILE  128          // points per workgroup
#define H1_PAD 520         // bf16 elements per h1 row (512 + 8 pad)
#define H2_PAD 268         // f32 elements per h2 row (256 + 12 pad)
#define EPSc  1e-5f
#define SLOPEc 0.2f

// ---------------------------------------------------------------------------
// Prep: W2 [E][H][Hh] f32  ->  W2^T bf16 [E][Hh][H]  (so B-fragment K is
// contiguous per output column; one 32B load per lane per WMMA)
// ---------------------------------------------------------------------------
__global__ void w2_transpose_bf16(const float* __restrict__ W2,
                                  bf16_t* __restrict__ w2t) {
    int idx = blockIdx.x * 256 + threadIdx.x;
    const int total = Ec * Hc * Hh;
    if (idx >= total) return;
    int e = idx / (Hc * Hh);
    int r = idx % (Hc * Hh);
    int n = r / Hc;          // output column 0..255
    int k = r % Hc;          // input feature 0..511
    w2t[idx] = (bf16_t)W2[(size_t)e * Hc * Hh + (size_t)k * Hh + n];
}

// ---------------------------------------------------------------------------
// Fused expert MLP: (3 -> 512) + LN + lrelu -> WMMA (512 -> 256) + LN + lrelu
// -> (256 -> 3)
// ---------------------------------------------------------------------------
__global__ __launch_bounds__(256, 1)
void fused_expert_mlp(const float* __restrict__ points,
                      const int*   __restrict__ cat,
                      const float* __restrict__ W1, const float* __restrict__ b1,
                      const float* __restrict__ g1, const float* __restrict__ be1,
                      const bf16_t* __restrict__ w2t, const float* __restrict__ b2,
                      const float* __restrict__ g2, const float* __restrict__ be2,
                      const float* __restrict__ W3, const float* __restrict__ b3,
                      float* __restrict__ out) {
    extern __shared__ char smem[];
    bf16_t* h1 = (bf16_t*)smem;                                   // [128][520]
    float*  h2 = (float*)(smem + TILE * H1_PAD * sizeof(bf16_t)); // [128][268]

    const int tiles_per_b = Nc / TILE;            // 64
    const int bidx = blockIdx.x / tiles_per_b;
    const int tile = blockIdx.x % tiles_per_b;
    const int p0   = tile * TILE;
    const int e    = cat[bidx];                   // uniform -> SGPR

    const int tid  = threadIdx.x;
    const int wave = tid >> 5;                    // 0..7
    const int lane = tid & 31;
    const int m    = lane >> 1;                   // point-in-wave 0..15
    const int s    = lane & 1;                    // half selector
    const int row  = wave * 16 + m;               // point row 0..127
    const int gp   = p0 + row;                    // point index in batch

    // ================= Phase A: layer1 + LN1 + lrelu -> bf16 LDS ===========
    const size_t pbase = ((size_t)bidx * Nc + gp) * 3;
    const float x = points[pbase + 0];
    const float y = points[pbase + 1];
    const float z = points[pbase + 2];
    const float* W1e = W1 + (size_t)e * 3 * Hc;
    const float* b1e = b1 + (size_t)e * Hc;

    float sum = 0.f, sumsq = 0.f;
    #pragma unroll 4
    for (int j = 0; j < 64; ++j) {
        const int f = j * 8 + s * 4;              // lane pair interleave
        v4f w0 = *(const v4f*)(W1e + 0 * Hc + f);
        v4f w1 = *(const v4f*)(W1e + 1 * Hc + f);
        v4f w2 = *(const v4f*)(W1e + 2 * Hc + f);
        v4f bb = *(const v4f*)(b1e + f);
        v4bf hv;
        #pragma unroll
        for (int i = 0; i < 4; ++i) {
            float h = fmaf(x, w0[i], fmaf(y, w1[i], fmaf(z, w2[i], bb[i])));
            sum += h; sumsq += h * h;
            hv[i] = (bf16_t)h;
        }
        *(v4bf*)(h1 + (size_t)row * H1_PAD + f) = hv;   // 8B aligned, bank-clean
    }
    sum   += __shfl_xor(sum, 1);
    sumsq += __shfl_xor(sumsq, 1);
    float mu  = sum * (1.0f / Hc);
    float var = sumsq * (1.0f / Hc) - mu * mu;
    float rs  = rsqrtf(var + EPSc);

    const float* g1e  = g1  + (size_t)e * Hc;
    const float* be1e = be1 + (size_t)e * Hc;
    #pragma unroll 4
    for (int j = 0; j < 64; ++j) {
        const int f = j * 8 + s * 4;
        v4bf hv = *(v4bf*)(h1 + (size_t)row * H1_PAD + f);
        v4f gg = *(const v4f*)(g1e + f);
        v4f bb = *(const v4f*)(be1e + f);
        #pragma unroll
        for (int i = 0; i < 4; ++i) {
            float v = (float)hv[i];
            v = (v - mu) * rs * gg[i] + bb[i];
            v = v >= 0.f ? v : SLOPEc * v;
            hv[i] = (bf16_t)v;
        }
        *(v4bf*)(h1 + (size_t)row * H1_PAD + f) = hv;
    }
    __syncthreads();

    // ================= Phase B: layer2 GEMM on WMMA bf16 ===================
    const int ln = lane & 15;                     // column / A-row within tile
    const int g  = lane >> 4;                     // half-wave group
    const bf16_t* w2te = w2t + (size_t)e * Hh * Hc;
    const bf16_t* arow = h1 + (size_t)(wave * 16 + ln) * H1_PAD;
    const float*  b2e  = b2 + (size_t)e * Hh;

    for (int half = 0; half < 2; ++half) {
        v8f acc[8] = {};                          // 8 N-tiles x 16x16 f32
        for (int kk = 0; kk < Hc; kk += 32) {
            // A fragment 16x32: 16 contiguous K per lane (two 8-elem chunks)
            v8bf c1 = *(const v8bf*)(arow + kk + 16 * g);
            v8bf c2 = *(const v8bf*)(arow + kk + 16 * g + 8);
            v16bf a;
            #pragma unroll
            for (int i = 0; i < 8; ++i) { a[i] = c1[i]; a[8 + i] = c2[i]; }
            #pragma unroll
            for (int t = 0; t < 8; ++t) {
                const int n0 = half * 128 + t * 16;
                // B fragment 32x16: lane = column, 16 contiguous K per lane
                v16bf bfr = *(const v16bf*)(w2te + (size_t)(n0 + ln) * Hc
                                                  + kk + 16 * g);
                acc[t] = __builtin_amdgcn_wmma_f32_16x16x32_bf16(
                            false, a, false, bfr, (short)0, acc[t],
                            false, false);
            }
        }
        #pragma unroll
        for (int t = 0; t < 8; ++t) {
            const int n0 = half * 128 + t * 16;
            const float bias = b2e[n0 + ln];
            float* hrow = h2 + (size_t)(wave * 16 + 8 * g) * H2_PAD + n0 + ln;
            #pragma unroll
            for (int v = 0; v < 8; ++v)           // D: M = 8*g + v, N = ln
                hrow[(size_t)v * H2_PAD] = acc[t][v] + bias;
        }
    }
    __syncthreads();

    // ============ Phase C: LN2 + lrelu fused into layer3 (256 -> 3) ========
    const float* hr = h2 + (size_t)row * H2_PAD + s * 128;
    float s2 = 0.f, q2 = 0.f;
    #pragma unroll 8
    for (int k = 0; k < 128; ++k) { float v = hr[k]; s2 += v; q2 += v * v; }
    s2 += __shfl_xor(s2, 1);
    q2 += __shfl_xor(q2, 1);
    mu  = s2 * (1.0f / Hh);
    var = q2 * (1.0f / Hh) - mu * mu;
    rs  = rsqrtf(var + EPSc);

    const float* g2e  = g2  + (size_t)e * Hh + s * 128;
    const float* be2e = be2 + (size_t)e * Hh + s * 128;
    const float* W3e  = W3  + (size_t)e * Hh * 3 + (size_t)s * 128 * 3;
    float o0 = 0.f, o1 = 0.f, o2 = 0.f;
    #pragma unroll 4
    for (int k = 0; k < 128; ++k) {
        float v = (hr[k] - mu) * rs * g2e[k] + be2e[k];
        v = v >= 0.f ? v : SLOPEc * v;
        o0 = fmaf(v, W3e[k * 3 + 0], o0);
        o1 = fmaf(v, W3e[k * 3 + 1], o1);
        o2 = fmaf(v, W3e[k * 3 + 2], o2);
    }
    o0 += __shfl_xor(o0, 1);
    o1 += __shfl_xor(o1, 1);
    o2 += __shfl_xor(o2, 1);
    if (s == 0) {
        const size_t ob = ((size_t)bidx * Nc + gp) * 3;
        out[ob + 0] = o0 + b3[e * 3 + 0];
        out[ob + 1] = o1 + b3[e * 3 + 1];
        out[ob + 2] = o2 + b3[e * 3 + 2];
    }
}

extern "C" void kernel_launch(void* const* d_in, const int* in_sizes, int n_in,
                              void* d_out, int out_size, void* d_ws, size_t ws_size,
                              hipStream_t stream) {
    const float* points = (const float*)d_in[0];
    const int*   cat    = (const int*)  d_in[1];
    const float* W1     = (const float*)d_in[2];
    const float* b1     = (const float*)d_in[3];
    const float* g1     = (const float*)d_in[4];
    const float* be1    = (const float*)d_in[5];
    const float* W2     = (const float*)d_in[6];
    const float* b2     = (const float*)d_in[7];
    const float* g2     = (const float*)d_in[8];
    const float* be2    = (const float*)d_in[9];
    const float* W3     = (const float*)d_in[10];
    const float* b3     = (const float*)d_in[11];
    float* out = (float*)d_out;

    bf16_t* w2t = (bf16_t*)d_ws;   // needs E*Hh*Hc*2 = 2.62 MB of scratch

    // 1) convert + transpose W2 to bf16 (tiny, once per launch)
    const int total = Ec * Hc * Hh;
    w2_transpose_bf16<<<(total + 255) / 256, 256, 0, stream>>>(W2, w2t);

    // 2) fused MLP
    const size_t lds_bytes = (size_t)TILE * H1_PAD * sizeof(bf16_t)
                           + (size_t)TILE * H2_PAD * sizeof(float);   // 270,336 B
    hipFuncSetAttribute(reinterpret_cast<const void*>(fused_expert_mlp),
                        hipFuncAttributeMaxDynamicSharedMemorySize,
                        (int)lds_bytes);
    const int nblocks = Bc * (Nc / TILE);   // 2048 workgroups
    fused_expert_mlp<<<nblocks, 256, lds_bytes, stream>>>(
        points, cat, W1, b1, g1, be1, w2t, b2, g2, be2, W3, b3, out);
}
